// MultiHeadAttention_17952963297409
// MI455X (gfx1250) — matrix operand order repaired and checked by reading, not compiled, not run on hardware
//
#include <hip/hip_runtime.h>
#include <hip/hip_bf16.h>

// ---------------------------------------------------------------------------
// MHA forward, B=4, L=1024, D=1024, H=16, HD=64 on gfx1250 (MI455X).
// bf16 WMMA (f32 accumulate) everywhere; async global->LDS tile staging.
// ---------------------------------------------------------------------------

#define B_  4
#define L_  1024
#define D_  1024
#define H_  16
#define HD_ 64
#define M_  (B_ * L_)          // 4096 rows
#define TS  64                 // N/K tile
#define MT  128                // GEMM M tile (32 rows per wave)
#define LSTR 72                // LDS row stride in bf16 elems (36 dwords, conflict-free)
#define NEGBIG (-1.0e9f)

typedef __attribute__((ext_vector_type(16))) __bf16 v16bf;
typedef __attribute__((ext_vector_type(8)))  float  v8f;
typedef __attribute__((ext_vector_type(4)))  int    v4i;

union BF16x16 { v16bf v; unsigned int u[8]; };

// Pointer types matching the gfx1250 async-load builtin prototype:
//   (v4i addrspace(1)* src, v4i addrspace(3)* dst, imm offset, imm cpol)
typedef __attribute__((address_space(1))) v4i* gbl_v4ip;
typedef __attribute__((address_space(3))) v4i* lds_v4ip;

__device__ __forceinline__ unsigned short f32_to_bf16(float f) {
    unsigned int u = __float_as_uint(f);
    u += 0x7FFFu + ((u >> 16) & 1u);     // round-to-nearest-even
    return (unsigned short)(u >> 16);
}

// 16-byte async global->LDS copy (ASYNCcnt-tracked on gfx1250).
__device__ __forceinline__ void async_b128(const void* g, void* l) {
#if __has_builtin(__builtin_amdgcn_global_load_async_to_lds_b128)
    __builtin_amdgcn_global_load_async_to_lds_b128((gbl_v4ip)g, (lds_v4ip)l, 0, 0);
#else
    *(uint4*)l = *(const uint4*)g;       // synchronous fallback
#endif
}

__device__ __forceinline__ void wait_async() {
#if __has_builtin(__builtin_amdgcn_s_wait_asynccnt)
    __builtin_amdgcn_s_wait_asynccnt(0);
#else
    asm volatile("s_wait_asynccnt 0" ::: "memory");
#endif
}

// A fragment (16x32 bf16): lane%16 = row; lanes 0-15 hold K {0..7,16..23},
// lanes 16-31 hold K {8..15,24..31} (ISA 7.12.2). rowbase = wave's 16-row
// sub-tile in LDS ([row][k], stride LSTR).
__device__ __forceinline__ v16bf load_a_frag(const unsigned short* rowbase,
                                             int kc, int hw, int l16) {
    BF16x16 f;
    const unsigned int* p =
        (const unsigned int*)(rowbase + (size_t)l16 * LSTR + kc + 8 * hw);
    f.u[0] = p[0]; f.u[1] = p[1]; f.u[2] = p[2]; f.u[3] = p[3];   // K +0..7
    f.u[4] = p[8]; f.u[5] = p[9]; f.u[6] = p[10]; f.u[7] = p[11]; // K +16..23
    return f.v;
}

// B fragment (32x16 bf16): lane%16 = col n; lanes 0-15 hold K {0..7,16..23}, lanes 16-31 hold K {8..15,24..31}.
// nbase points at a [n][k] (transposed) LDS tile sub-block, stride LSTR.
__device__ __forceinline__ v16bf load_b_frag(const unsigned short* nbase,
                                             int kc, int hw, int l16) {
    BF16x16 f;
    const unsigned int* p =
        (const unsigned int*)(nbase + (size_t)l16 * LSTR + kc + 8 * hw);
#pragma unroll
    for (int j = 0; j < 8; ++j) f.u[j] = p[j < 4 ? j : j + 4];
    return f.v;
}

__device__ __forceinline__ v8f wmma_bf16(v16bf a, v16bf b, v8f c) {
    return __builtin_amdgcn_wmma_f32_16x16x32_bf16(
        false, a, false, b, (short)0, c, false, false);
}

// ---------------------------------------------------------------------------
// Precast kernels: f32 -> bf16 (and transposed variant for weights).
// ---------------------------------------------------------------------------
__global__ void __launch_bounds__(256)
cast_bf16(const float* __restrict__ s, unsigned short* __restrict__ d, int n) {
    int i = (blockIdx.x * 256 + threadIdx.x) * 4;
    if (i >= n) return;
    float4 f = *(const float4*)(s + i);
    unsigned int lo = (unsigned int)f32_to_bf16(f.x) |
                      ((unsigned int)f32_to_bf16(f.y) << 16);
    unsigned int hi = (unsigned int)f32_to_bf16(f.z) |
                      ((unsigned int)f32_to_bf16(f.w) << 16);
    *(uint2*)(d + i) = make_uint2(lo, hi);
}

// W [k][n] f32  ->  WT [n][k] bf16 via LDS tile (coalesced both sides).
__global__ void __launch_bounds__(256)
cast_transpose(const float* __restrict__ W, unsigned short* __restrict__ WT) {
    __shared__ unsigned short t[32][33];
    const int bn = blockIdx.x * 32, bk = blockIdx.y * 32;
    const int tx = threadIdx.x & 31, ty = threadIdx.x >> 5;   // 32 x 8
#pragma unroll
    for (int i = 0; i < 32; i += 8)
        t[ty + i][tx] = f32_to_bf16(W[(size_t)(bk + ty + i) * D_ + bn + tx]); // t[k][n]
    __syncthreads();
#pragma unroll
    for (int i = 0; i < 32; i += 8)
        WT[(size_t)(bn + ty + i) * D_ + bk + tx] = t[tx][ty + i];
}

// ---------------------------------------------------------------------------
// GEMM: Out = A[M,1024](bf16) @ W, W given pre-transposed WT[n][k] bf16.
// Block 128 thr / 4 waves; tile 128x64; each wave owns 32 rows.
// OUT_MODE 0: bf16 head-split [B,H,L,HD]
// OUT_MODE 1: bf16 head-split transposed [B,H,HD,L]   (for V)
// OUT_MODE 2: f32 plain [M,D]
// ---------------------------------------------------------------------------
template <int OUT_MODE>
__global__ void __launch_bounds__(128)
gemm_wmma(const unsigned short* __restrict__ A,
          const unsigned short* __restrict__ WT,
          void* __restrict__ Out) {
    __shared__ unsigned short Xs[MT * LSTR];   // [m][k]
    __shared__ unsigned short Wt[TS * LSTR];   // [n][k]

    const int tid = threadIdx.x;
    const int lane = tid & 31, wv = tid >> 5;
    const int hw = lane >> 4, l16 = lane & 15;
    const int m0 = blockIdx.x * MT;
    const int n0 = blockIdx.y * TS;

    v8f acc[2][4] = {};   // 2 M-subtiles x 4 N-tiles of 16x16 f32

    for (int kb = 0; kb < D_; kb += TS) {
        __syncthreads();
        // ---- async stage A tile [128][64] ----
#pragma unroll
        for (int i = 0; i < 8; ++i) {
            int e = (tid + i * 128) * 8;
            int r = e >> 6, c = e & 63;
            async_b128(A + (size_t)(m0 + r) * D_ + kb + c, &Xs[r * LSTR + c]);
        }
        // ---- async stage W tile [64 n][64 k] from WT[n][k] ----
#pragma unroll
        for (int i = 0; i < 4; ++i) {
            int e = (tid + i * 128) * 8;
            int r = e >> 6, c = e & 63;
            async_b128(WT + (size_t)(n0 + r) * D_ + kb + c, &Wt[r * LSTR + c]);
        }
        wait_async();
        __syncthreads();

        // ---- compute: wave owns rows [32*wv, 32*wv+32) ----
#pragma unroll
        for (int kc = 0; kc < TS; kc += 32) {
            v16bf a0 = load_a_frag(&Xs[(wv * 32) * LSTR], kc, hw, l16);
            v16bf a1 = load_a_frag(&Xs[(wv * 32 + 16) * LSTR], kc, hw, l16);
#pragma unroll
            for (int nt = 0; nt < 4; ++nt) {
                v16bf b = load_b_frag(&Wt[nt * 16 * LSTR], kc, hw, l16);
                acc[0][nt] = wmma_bf16(a0, b, acc[0][nt]);
                acc[1][nt] = wmma_bf16(a1, b, acc[1][nt]);
            }
        }
    }

    // ---- store C (m = r + 8*hw, n = l16 within each 16x16 tile) ----
#pragma unroll
    for (int mt = 0; mt < 2; ++mt) {
#pragma unroll
        for (int nt = 0; nt < 4; ++nt) {
#pragma unroll
            for (int r = 0; r < 8; ++r) {
                int m = m0 + wv * 32 + mt * 16 + r + 8 * hw;
                int n = n0 + nt * 16 + l16;
                float val = acc[mt][nt][r];
                if constexpr (OUT_MODE == 0) {
                    int b = m >> 10, l = m & (L_ - 1);
                    int h = n >> 6,  hd = n & (HD_ - 1);
                    ((unsigned short*)Out)[((((size_t)(b * H_ + h)) << 10) + l) * HD_ + hd] =
                        f32_to_bf16(val);
                } else if constexpr (OUT_MODE == 1) {
                    int b = m >> 10, l = m & (L_ - 1);
                    int h = n >> 6,  hd = n & (HD_ - 1);
                    ((unsigned short*)Out)[((size_t)(b * H_ + h) * HD_ + hd) * L_ + l] =
                        f32_to_bf16(val);
                } else {
                    ((float*)Out)[(size_t)m * D_ + n] = val;
                }
            }
        }
    }
}

// ---------------------------------------------------------------------------
// Flash attention: one block per (b, h, 64-query tile). 4 waves; each wave
// owns 16 query rows. Online softmax over 64-key blocks.
// Qh/Kh: [B,H,L,HD] bf16.  VhT: [B,H,HD,L] bf16 (pre-transposed).
// ---------------------------------------------------------------------------
__global__ void __launch_bounds__(128)
flash_attn(const unsigned short* __restrict__ Qh,
           const unsigned short* __restrict__ Kh,
           const unsigned short* __restrict__ VhT,
           const float* __restrict__ mask,
           const int* __restrict__ valid_lens,
           unsigned short* __restrict__ attn /* [M, D] bf16 */) {
    __shared__ unsigned short Qs[TS * LSTR];        // [q][hd]
    __shared__ unsigned short Ks[TS * LSTR];        // [key][hd] == B^T form for QK^T
    __shared__ unsigned short Vt[TS * LSTR];        // [hd][key]
    __shared__ unsigned short Ps[4 * 16 * LSTR];    // per-wave P [16 q][64 key]

    const int tid = threadIdx.x;
    const int lane = tid & 31, wv = tid >> 5;
    const int hw = lane >> 4, l16 = lane & 15;
    const int bh = blockIdx.x >> 4;                 // 16 q-tiles per (b,h)
    const int qb = (blockIdx.x & 15) * TS;
    const int b = bh >> 4;                          // H = 16
    const int h = bh & 15;
    const int vl = valid_lens[b];

    const unsigned short* Qg = Qh  + (size_t)bh * L_ * HD_;
    const unsigned short* Kg = Kh  + (size_t)bh * L_ * HD_;
    const unsigned short* Vg = VhT + (size_t)bh * HD_ * L_;

    // ---- async stage Q tile once ----
#pragma unroll
    for (int i = 0; i < 4; ++i) {
        int e = (tid + i * 128) * 8;
        int r = e >> 6, c = e & 63;
        async_b128(Qg + (size_t)(qb + r) * HD_ + c, &Qs[r * LSTR + c]);
    }
    wait_async();
    __syncthreads();
    v16bf qa0 = load_a_frag(&Qs[wv * 16 * LSTR], 0, hw, l16);
    v16bf qa1 = load_a_frag(&Qs[wv * 16 * LSTR], 32, hw, l16);

    float mi[8], li[8];
    v8f O[4] = {};
#pragma unroll
    for (int r = 0; r < 8; ++r) { mi[r] = -1.0e30f; li[r] = 0.0f; }

    const int qrow0 = qb + wv * 16 + 8 * hw;   // + r gives this lane's query row
    const float scale = 0.125f;                // 1/sqrt(HD)

    for (int kb = 0; kb < L_; kb += TS) {
        __syncthreads();
        // ---- async stage K (row-major) and V (already transposed) ----
#pragma unroll
        for (int i = 0; i < 4; ++i) {
            int e = (tid + i * 128) * 8;
            int r = e >> 6, c = e & 63;
            async_b128(Kg + (size_t)(kb + r) * HD_ + c, &Ks[r * LSTR + c]);
            async_b128(Vg + (size_t)r * L_ + kb + c,    &Vt[r * LSTR + c]);
        }
        wait_async();
        __syncthreads();

        // ---- S tile = Q K^T (scaled) + mask + symmetric valid mask ----
        float s[4][8];
#pragma unroll
        for (int nt = 0; nt < 4; ++nt) {
            v8f a = {};
            a = wmma_bf16(qa0, load_b_frag(&Ks[nt * 16 * LSTR], 0, hw, l16), a);
            a = wmma_bf16(qa1, load_b_frag(&Ks[nt * 16 * LSTR], 32, hw, l16), a);
            int kcol = kb + nt * 16 + l16;
            bool kmask = (kcol >= vl);
#pragma unroll
            for (int r = 0; r < 8; ++r) {
                int qrow = qrow0 + r;
                float sv = a[r] * scale + mask[(size_t)qrow * L_ + kcol];
                if (kmask || (qrow >= vl)) sv += NEGBIG;
                s[nt][r] = sv;
            }
        }

        // ---- online softmax update (per row r, across 16 lanes) ----
        float corr[8];
#pragma unroll
        for (int r = 0; r < 8; ++r) {
            float v = fmaxf(fmaxf(s[0][r], s[1][r]), fmaxf(s[2][r], s[3][r]));
            v = fmaxf(v, __shfl_xor(v, 1, 32));
            v = fmaxf(v, __shfl_xor(v, 2, 32));
            v = fmaxf(v, __shfl_xor(v, 4, 32));
            v = fmaxf(v, __shfl_xor(v, 8, 32));
            float mnew = fmaxf(mi[r], v);
            corr[r] = __expf(mi[r] - mnew);
            mi[r] = mnew;
        }

        unsigned short* Pw = &Ps[wv * 16 * LSTR];
        float rs[8] = {0, 0, 0, 0, 0, 0, 0, 0};
#pragma unroll
        for (int nt = 0; nt < 4; ++nt) {
#pragma unroll
            for (int r = 0; r < 8; ++r) {
                float p = __expf(s[nt][r] - mi[r]);
                rs[r] += p;
                Pw[(r + 8 * hw) * LSTR + nt * 16 + l16] = f32_to_bf16(p);
            }
        }
#pragma unroll
        for (int r = 0; r < 8; ++r) {
            float v = rs[r];
            v += __shfl_xor(v, 1, 32);
            v += __shfl_xor(v, 2, 32);
            v += __shfl_xor(v, 4, 32);
            v += __shfl_xor(v, 8, 32);
            li[r] = li[r] * corr[r] + v;
        }
#pragma unroll
        for (int ht = 0; ht < 4; ++ht)
#pragma unroll
            for (int r = 0; r < 8; ++r) O[ht][r] *= corr[r];

        // cross-lane LDS dependency inside the wave: drain DS before re-reading
        asm volatile("s_wait_dscnt 0" ::: "memory");

        // ---- O += P @ V ----
#pragma unroll
        for (int kc = 0; kc < TS; kc += 32) {
            v16bf pa = load_a_frag(Pw, kc, hw, l16);
#pragma unroll
            for (int ht = 0; ht < 4; ++ht) {
                v16bf bv = load_b_frag(&Vt[ht * 16 * LSTR], kc, hw, l16);
                O[ht] = wmma_bf16(pa, bv, O[ht]);
            }
        }
    }

    // ---- normalize, store bf16 into attn [B*L, D] ----
#pragma unroll
    for (int ht = 0; ht < 4; ++ht) {
#pragma unroll
        for (int r = 0; r < 8; ++r) {
            int qrow = qrow0 + r;
            int col = h * HD_ + ht * 16 + l16;
            attn[(size_t)(b * L_ + qrow) * D_ + col] =
                f32_to_bf16(O[ht][r] / li[r]);
        }
    }
}

// ---------------------------------------------------------------------------
// Host-side launch. Workspace (bf16 elements):
//   qb,kb,vb [M,D]  wqT,wkT,wvT,woT [D,D]  Qh,Kh [B,H,L,HD]  VhT [B,H,HD,L]
//   attn aliases qb (q no longer needed once Qh exists).
// ---------------------------------------------------------------------------
extern "C" void kernel_launch(void* const* d_in, const int* in_sizes, int n_in,
                              void* d_out, int out_size, void* d_ws, size_t ws_size,
                              hipStream_t stream) {
    (void)in_sizes; (void)n_in; (void)out_size; (void)ws_size;

    const float* q   = (const float*)d_in[0];
    const float* k   = (const float*)d_in[1];
    const float* v   = (const float*)d_in[2];
    const float* wq  = (const float*)d_in[3];
    const float* wk  = (const float*)d_in[4];
    const float* wv  = (const float*)d_in[5];
    const float* wo  = (const float*)d_in[6];
    const float* msk = (const float*)d_in[7];
    const int*   vls = (const int*)d_in[8];

    const size_t mdElems = (size_t)M_ * D_;   // 4 Mi
    const size_t ddElems = (size_t)D_ * D_;   // 1 Mi
    unsigned short* p    = (unsigned short*)d_ws;
    unsigned short* qb   = p; p += mdElems;
    unsigned short* kb   = p; p += mdElems;
    unsigned short* vb   = p; p += mdElems;
    unsigned short* wqT  = p; p += ddElems;
    unsigned short* wkT  = p; p += ddElems;
    unsigned short* wvT  = p; p += ddElems;
    unsigned short* woT  = p; p += ddElems;
    unsigned short* Qh   = p; p += mdElems;
    unsigned short* Kh   = p; p += mdElems;
    unsigned short* VhT  = p; p += mdElems;
    unsigned short* attn = qb;                // alias: q consumed before attention

    dim3 blk(128);
    dim3 gCast((unsigned)(mdElems / 4 / 256));
    dim3 gCastT(D_ / 32, D_ / 32);
    dim3 gGemm(M_ / MT, D_ / TS);             // (32, 16)

    // f32 -> bf16 precasts (enable pure-bf16 async staging everywhere)
    cast_bf16<<<gCast, 256, 0, stream>>>(q, qb, (int)mdElems);
    cast_bf16<<<gCast, 256, 0, stream>>>(k, kb, (int)mdElems);
    cast_bf16<<<gCast, 256, 0, stream>>>(v, vb, (int)mdElems);
    cast_transpose<<<gCastT, 256, 0, stream>>>(wq, wqT);
    cast_transpose<<<gCastT, 256, 0, stream>>>(wk, wkT);
    cast_transpose<<<gCastT, 256, 0, stream>>>(wv, wvT);
    cast_transpose<<<gCastT, 256, 0, stream>>>(wo, woT);

    // Projections
    gemm_wmma<0><<<gGemm, blk, 0, stream>>>(qb, wqT, Qh);
    gemm_wmma<0><<<gGemm, blk, 0, stream>>>(kb, wkT, Kh);
    gemm_wmma<1><<<gGemm, blk, 0, stream>>>(vb, wvT, VhT);

    // Flash attention: B*H*(L/64) = 1024 blocks
    flash_attn<<<dim3(B_ * H_ * (L_ / TS)), blk, 0, stream>>>(
        Qh, Kh, VhT, msk, vls, attn);

    // Output projection (bf16 in, f32 out)
    gemm_wmma<2><<<gGemm, blk, 0, stream>>>(attn, woT, (float*)d_out);
}
